// Attention_79070347919638
// MI455X (gfx1250) — matrix operand order repaired and checked by reading, not compiled, not run on hardware
//
#include <hip/hip_runtime.h>
#include <math.h>

// ---------------- CDNA5 WMMA types ----------------
typedef __bf16 v16bf __attribute__((ext_vector_type(16)));
typedef float  v8f   __attribute__((ext_vector_type(8)));
typedef unsigned int u32x4 __attribute__((ext_vector_type(4)));

static constexpr int Bb  = 16;
static constexpr int Nn  = 2048;
static constexpr int Dd  = 256;
static constexpr int Ii  = 512;    // H*DH
static constexpr int Hh  = 8;
static constexpr int DHh = 64;
static constexpr int Kk  = 8;
static constexpr int ROWS = Bb * Nn;      // 32768
static constexpr int QKVC = 3 * Ii;       // 1536

// ---------------- helpers ----------------
__device__ __forceinline__ unsigned short f2bf(float f) {
    unsigned u = __float_as_uint(f);
    unsigned r = u + 0x7FFFu + ((u >> 16) & 1u);   // round-to-nearest-even
    return (unsigned short)(r >> 16);
}
__device__ __forceinline__ float bf_lo(unsigned u) { return __uint_as_float(u << 16); }
__device__ __forceinline__ float bf_hi(unsigned u) { return __uint_as_float(u & 0xFFFF0000u); }

union FragBF { u32x4 u[2]; v16bf v; };

// ---------------- weight transpose + fp32->bf16 convert ----------------
// Wt[n][k] = bf16(W[k][n]);  W is [Krows x Ncols] row-major, Wt is [Ncols x Krows]
__global__ void tr_cvt_bf16_kernel(const float* __restrict__ W,
                                   unsigned short* __restrict__ Wt,
                                   int Krows, int Ncols) {
    int i = blockIdx.x * blockDim.x + threadIdx.x;
    if (i >= Krows * Ncols) return;
    int n = i / Krows, k = i - n * Krows;
    Wt[(size_t)n * Krows + k] = f2bf(W[(size_t)k * Ncols + n]);
}

// ---------------- ball query: first K indices (ascending) with d2 < 0.09 ----------------
__global__ __launch_bounds__(128) void ball_query_kernel(const float* __restrict__ xyz,
                                                         int* __restrict__ idxout) {
    __shared__ float sx[128], sy[128], sz[128];
    int b = blockIdx.y;
    int n = blockIdx.x * 128 + threadIdx.x;
    const float* base = xyz + (size_t)b * Nn * 3;
    float px = base[n * 3 + 0], py = base[n * 3 + 1], pz = base[n * 3 + 2];
    int nb[Kk];
    int cnt = 0;
#pragma unroll
    for (int i = 0; i < Kk; i++) nb[i] = 0;
    for (int mt = 0; mt < Nn; mt += 128) {
        __syncthreads();
        int m = mt + threadIdx.x;
        sx[threadIdx.x] = base[m * 3 + 0];
        sy[threadIdx.x] = base[m * 3 + 1];
        sz[threadIdx.x] = base[m * 3 + 2];
        __syncthreads();
        for (int j = 0; j < 128; j++) {
            float dx = sx[j] - px, dy = sy[j] - py, dz = sz[j] - pz;
            float d2 = dx * dx + dy * dy + dz * dz;
            if (d2 < 0.09f && cnt < Kk) {
#pragma unroll
                for (int i = 0; i < Kk; i++) if (cnt == i) nb[i] = mt + j;
                cnt++;
            }
        }
    }
#pragma unroll
    for (int i = 0; i < Kk; i++) if (i >= cnt) nb[i] = nb[0];   // self always hits -> cnt>=1
    int* o = idxout + ((size_t)b * Nn + n) * Kk;
#pragma unroll
    for (int i = 0; i < Kk; i++) o[i] = nb[i];
}

// ---------------- LayerNorm -> bf16 (wave per row, 8 waves per block) ----------------
__global__ __launch_bounds__(256) void layernorm_kernel(const float* __restrict__ x,
                                                        const float* __restrict__ g,
                                                        const float* __restrict__ be,
                                                        unsigned short* __restrict__ ob) {
    int lane = threadIdx.x & 31, w = threadIdx.x >> 5;
    size_t row = (size_t)blockIdx.x * 8 + w;
    const float* xr = x + row * Dd;
    float v[8], s1 = 0.f, s2 = 0.f;
#pragma unroll
    for (int i = 0; i < 8; i++) { v[i] = xr[lane + 32 * i]; s1 += v[i]; s2 += v[i] * v[i]; }
#pragma unroll
    for (int off = 16; off >= 1; off >>= 1) {
        s1 += __shfl_xor(s1, off, 32);
        s2 += __shfl_xor(s2, off, 32);
    }
    float mu  = s1 * (1.0f / Dd);
    float var = s2 * (1.0f / Dd) - mu * mu;
    float rs  = rsqrtf(var + 1e-5f);
#pragma unroll
    for (int i = 0; i < 8; i++) {
        int c = lane + 32 * i;
        ob[row * Dd + c] = f2bf((v[i] - mu) * rs * g[c] + be[c]);
    }
}

// ---------------- WMMA GEMM: qkv = normed[ROWS x 256] @ wqT^T, bf16 out ----------------
// A row-major bf16 [ROWS][256]; Bt column-major bf16 [1536][256]; C bf16 [ROWS][1536]
__global__ __launch_bounds__(256) void gemm_qkv_kernel(const unsigned short* __restrict__ A,
                                                       const unsigned short* __restrict__ Bt,
                                                       unsigned short* __restrict__ C) {
    constexpr int KD = Dd, NC = QKVC;
    int lane = threadIdx.x & 31, w = threadIdx.x >> 5;
    int half = lane >> 4, l15 = lane & 15;
    int rowBase = blockIdx.x * 128 + w * 16;
    int colBase = blockIdx.y * 128;
    v8f acc[8] = {};
    for (int kt = 0; kt < KD / 32; ++kt) {
        FragBF a;
        const unsigned short* ap = A + (size_t)(rowBase + l15) * KD + kt * 32 + 16 * half;
        a.u[0] = *(const u32x4*)(ap);
        a.u[1] = *(const u32x4*)(ap + 8);
#pragma unroll
        for (int t = 0; t < 8; t++) {
            FragBF bfr;
            const unsigned short* bp =
                Bt + (size_t)(colBase + t * 16 + l15) * KD + kt * 32 + 16 * half;
            bfr.u[0] = *(const u32x4*)(bp);
            bfr.u[1] = *(const u32x4*)(bp + 8);
            acc[t] = __builtin_amdgcn_wmma_f32_16x16x32_bf16(
                false, a.v, false, bfr.v, (short)0, acc[t], false, false);
        }
    }
#pragma unroll
    for (int t = 0; t < 8; t++) {
        int col = colBase + t * 16 + l15;
#pragma unroll
        for (int r = 0; r < 8; r++) {
            int row = rowBase + r + 8 * half;
            C[(size_t)row * NC + col] = f2bf(acc[t][r]);
        }
    }
}

// ---------------- neighborhood attention: wave per (b,n,head) ----------------
__global__ __launch_bounds__(256) void attn_kernel(const unsigned short* __restrict__ qkv,
                                                   const int* __restrict__ idx,
                                                   const float* __restrict__ xyz,
                                                   const float* __restrict__ wsp,
                                                   unsigned short* __restrict__ aod) {
    __shared__ int   sidx[Kk];
    __shared__ float sdisp[Kk][3];
    int b = blockIdx.y, n = blockIdx.x;
    int tid = threadIdx.x, lane = tid & 31, h = tid >> 5;
    if (tid < Kk) sidx[tid] = idx[((size_t)b * Nn + n) * Kk + tid];
    __syncthreads();
    if (tid < Kk * 3) {
        int k = tid / 3, c = tid - 3 * k;
        const float* xb = xyz + (size_t)b * Nn * 3;
        sdisp[k][c] = xb[(size_t)sidx[k] * 3 + c] - xb[(size_t)n * 3 + c];
    }
    __syncthreads();

    size_t rowq = ((size_t)b * Nn + n) * QKVC;
    unsigned qu = *(const unsigned*)(qkv + rowq + h * DHh + 2 * lane);
    float q0 = bf_lo(qu), q1 = bf_hi(qu);

    float lg[Kk];
#pragma unroll
    for (int k = 0; k < Kk; k++) {
        size_t rk = ((size_t)b * Nn + sidx[k]) * QKVC + Ii + h * DHh + 2 * lane;
        unsigned ku = *(const unsigned*)(qkv + rk);
        float p = q0 * bf_lo(ku) + q1 * bf_hi(ku);
#pragma unroll
        for (int off = 16; off >= 1; off >>= 1) p += __shfl_xor(p, off, 32);
        lg[k] = p * 0.125f;                                  // * DH^-0.5
    }
    float mx = lg[0];
#pragma unroll
    for (int k = 1; k < Kk; k++) mx = fmaxf(mx, lg[k]);
    float at[Kk], s = 0.f;
#pragma unroll
    for (int k = 0; k < Kk; k++) { at[k] = expf(lg[k] - mx); s += at[k]; }
    float inv = 1.0f / s;
#pragma unroll
    for (int k = 0; k < Kk; k++) at[k] *= inv;

    float a0 = 0.f, a1 = 0.f;
    float sp0 = at[0] * sdisp[0][0], sp1 = at[0] * sdisp[0][1], sp2 = at[0] * sdisp[0][2];
#pragma unroll
    for (int k = 0; k < Kk; k++) {
        size_t rv = ((size_t)b * Nn + sidx[k]) * QKVC + 2 * Ii + h * DHh + 2 * lane;
        unsigned vu = *(const unsigned*)(qkv + rv);
        a0 += at[k] * bf_lo(vu);
        a1 += at[k] * bf_hi(vu);
        if (k > 0) {
            sp0 = fmaxf(sp0, at[k] * sdisp[k][0]);
            sp1 = fmaxf(sp1, at[k] * sdisp[k][1]);
            sp2 = fmaxf(sp2, at[k] * sdisp[k][2]);
        }
    }
    int d0 = 2 * lane, d1 = d0 + 1;
    float dis0 = sp0 * wsp[d0] + sp1 * wsp[DHh + d0] + sp2 * wsp[2 * DHh + d0];
    float dis1 = sp0 * wsp[d1] + sp1 * wsp[DHh + d1] + sp2 * wsp[2 * DHh + d1];
    unsigned pack = (unsigned)f2bf(a0 + dis0) | ((unsigned)f2bf(a1 + dis1) << 16);
    *(unsigned*)(aod + ((size_t)b * Nn + n) * Ii + h * DHh + 2 * lane) = pack;
}

// ---------------- WMMA GEMM: out = gelu(aod @ wo + b) + feature (fp32 out) ----------------
// A bf16 [ROWS][512]; Bt bf16 column-major [256][512]
__global__ __launch_bounds__(256) void gemm_out_kernel(const unsigned short* __restrict__ A,
                                                       const unsigned short* __restrict__ Bt,
                                                       const float* __restrict__ bias,
                                                       const float* __restrict__ feat,
                                                       float* __restrict__ out) {
    constexpr int KD = Ii, NC = Dd;
    int lane = threadIdx.x & 31, w = threadIdx.x >> 5;
    int half = lane >> 4, l15 = lane & 15;
    int rowBase = blockIdx.x * 128 + w * 16;
    int colBase = blockIdx.y * 128;
    v8f acc[8] = {};
    for (int kt = 0; kt < KD / 32; ++kt) {
        FragBF a;
        const unsigned short* ap = A + (size_t)(rowBase + l15) * KD + kt * 32 + 16 * half;
        a.u[0] = *(const u32x4*)(ap);
        a.u[1] = *(const u32x4*)(ap + 8);
#pragma unroll
        for (int t = 0; t < 8; t++) {
            FragBF bfr;
            const unsigned short* bp =
                Bt + (size_t)(colBase + t * 16 + l15) * KD + kt * 32 + 16 * half;
            bfr.u[0] = *(const u32x4*)(bp);
            bfr.u[1] = *(const u32x4*)(bp + 8);
            acc[t] = __builtin_amdgcn_wmma_f32_16x16x32_bf16(
                false, a.v, false, bfr.v, (short)0, acc[t], false, false);
        }
    }
#pragma unroll
    for (int t = 0; t < 8; t++) {
        int col = colBase + t * 16 + l15;
        float bv = bias[col];
#pragma unroll
        for (int r = 0; r < 8; r++) {
            int row = rowBase + r + 8 * half;
            float x = acc[t][r] + bv;
            float gelu = 0.5f * x * (1.0f + erff(x * 0.70710678118654752f));  // exact gelu
            size_t o = (size_t)row * NC + col;
            out[o] = gelu + feat[o];
        }
    }
}

// ---------------- host launcher ----------------
extern "C" void kernel_launch(void* const* d_in, const int* in_sizes, int n_in,
                              void* d_out, int out_size, void* d_ws, size_t ws_size,
                              hipStream_t stream) {
    const float* xyzs    = (const float*)d_in[0];
    const float* feature = (const float*)d_in[1];
    const float* ln_g    = (const float*)d_in[2];
    const float* ln_b    = (const float*)d_in[3];
    const float* w_qkv   = (const float*)d_in[4];
    const float* w_sp    = (const float*)d_in[5];
    const float* w_out   = (const float*)d_in[6];
    const float* b_out   = (const float*)d_in[7];
    float* out = (float*)d_out;
    char* ws = (char*)d_ws;

    size_t o_idx  = 0;
    size_t o_norm = o_idx  + (size_t)ROWS * Kk * sizeof(int);        //  1.0 MB
    size_t o_wqT  = o_norm + (size_t)ROWS * Dd * 2;                  // 16.8 MB
    size_t o_woT  = o_wqT  + (size_t)QKVC * Dd * 2;                  //  0.8 MB
    size_t o_qkv  = o_woT  + (size_t)Dd * Ii * 2;                    //  0.26 MB
    size_t o_aod  = o_qkv  + (size_t)ROWS * QKVC * 2;                // 100.7 MB
    (void)ws_size; (void)in_sizes; (void)n_in; (void)out_size;

    int*            idxb  = (int*)(ws + o_idx);
    unsigned short* normb = (unsigned short*)(ws + o_norm);
    unsigned short* wqT   = (unsigned short*)(ws + o_wqT);
    unsigned short* woT   = (unsigned short*)(ws + o_woT);
    unsigned short* qkvb  = (unsigned short*)(ws + o_qkv);
    unsigned short* aodb  = (unsigned short*)(ws + o_aod);

    // 1) weight transpose + bf16 convert
    tr_cvt_bf16_kernel<<<(Dd * QKVC + 255) / 256, 256, 0, stream>>>(w_qkv, wqT, Dd, QKVC);
    tr_cvt_bf16_kernel<<<(Ii * Dd + 255) / 256, 256, 0, stream>>>(w_out, woT, Ii, Dd);
    // 2) ball query
    ball_query_kernel<<<dim3(Nn / 128, Bb), 128, 0, stream>>>(xyzs, idxb);
    // 3) LayerNorm -> bf16
    layernorm_kernel<<<ROWS / 8, 256, 0, stream>>>(feature, ln_g, ln_b, normb);
    // 4) QKV GEMM (WMMA bf16)
    gemm_qkv_kernel<<<dim3(ROWS / 128, QKVC / 128), 256, 0, stream>>>(normb, wqT, qkvb);
    // 5) neighborhood attention + spatial op -> bf16 aod
    attn_kernel<<<dim3(Nn, Bb), 256, 0, stream>>>(qkvb, idxb, xyzs, w_sp, aodb);
    // 6) output GEMM (WMMA bf16) + bias + exact GELU + residual
    gemm_out_kernel<<<dim3(ROWS / 128, Dd / 128), 256, 0, stream>>>(aodb, woT, b_out, feature, out);
}
